// MultiHeadDeformableAttention_84756884619497
// MI455X (gfx1250) — matrix operand order repaired and checked by reading, not compiled, not run on hardware
//
#include <hip/hip_runtime.h>
#include <hip/hip_bf16.h>

typedef __attribute__((ext_vector_type(16))) __bf16          v16bf;
typedef __attribute__((ext_vector_type(16))) unsigned short  v16u16;
typedef __attribute__((ext_vector_type(8)))  float           v8f;

#define QTOT  13294
#define BATCH 2
#define EDIM  256
#define HEADS 8
#define MROWS (BATCH * QTOT)   /* 26588 rows == B*Q == B*num_features */

__device__ __forceinline__ unsigned short f2bf(float f) {
    union { float f; unsigned int u; } v; v.f = f;
    unsigned int u = v.u;
    u += 0x7FFFu + ((u >> 16) & 1u);          // round-to-nearest-even
    return (unsigned short)(u >> 16);
}
__device__ __forceinline__ unsigned int pack2bf(float x, float y) {
    return (unsigned int)f2bf(x) | ((unsigned int)f2bf(y) << 16);
}
__device__ __forceinline__ float bf2f(unsigned short u) {
    union { unsigned int i; float f; } v; v.i = (unsigned int)u << 16;
    return v.f;
}

// ---------------------------------------------------------------------------
// Bulk fp32 -> bf16 conversion, 8 elements per thread (memory bound).
// ---------------------------------------------------------------------------
__global__ __launch_bounds__(256)
void f32_to_bf16_kernel(const float* __restrict__ src,
                        unsigned short* __restrict__ dst, int n8) {
    const int i = blockIdx.x * 256 + threadIdx.x;
    if (i >= n8) return;
    const float4* s = (const float4*)src + (size_t)i * 2;
    const float4 a = s[0];
    const float4 b = s[1];
    uint4 o;
    o.x = pack2bf(a.x, a.y);
    o.y = pack2bf(a.z, a.w);
    o.z = pack2bf(b.x, b.y);
    o.w = pack2bf(b.z, b.w);
    ((uint4*)dst)[i] = o;
}

// ---------------------------------------------------------------------------
// C[M,N] = A_bf16[M,256] x bf16(W[256,N]) + bias[N], f32 accumulate.
// Block: 256 threads = 8 waves; block tile 128 rows x 64 cols.
// Weights staged once per block into LDS in exact per-lane B-fragment layout.
// Inner loop: clause-load all 4 B fragments + A fragment, single dscnt wait,
// then 4 back-to-back WMMAs on independent accumulators.
// ---------------------------------------------------------------------------
template <bool OUT_BF16>
__global__ __launch_bounds__(256)
void gemm_bf16_wmma(const unsigned short* __restrict__ A,
                    const float* __restrict__ W,
                    const float* __restrict__ bias,
                    void* __restrict__ Cv, int M, int N) {
    __shared__ __align__(32) unsigned short ldsB[4][8][32][16];  // 32 KB

    const int tid   = threadIdx.x;
    const int lane  = tid & 31;
    const int wid   = tid >> 5;
    const int nblk  = blockIdx.y * 64;
    const int lhalf = (lane >= 16) ? 1 : 0;
    const int lcol  = lane & 15;

    // Cooperative bf16 conversion of the 256x64 weight slab into LDS,
    // pre-swizzled into B-fragment layout (lanes 0-15: K 0..7,16..23; 16-31: K 8..15,24..31).
    for (int fid = wid; fid < 32; fid += 8) {
        const int nt = fid >> 3;           // n-tile within block (0..3)
        const int s  = fid & 7;            // k-step (0..7)
        const int n  = nblk + nt * 16 + lcol;
        const int kb = s * 32 + (lhalf ? 8 : 0);
        #pragma unroll
        for (int e = 0; e < 16; ++e)
            ldsB[nt][s][lane][e] = f2bf(W[(size_t)(kb + (e < 8 ? e : e + 8)) * N + n]);
    }
    __syncthreads();

    const int mbase = (blockIdx.x * 8 + wid) * 16;
    if (mbase >= M) return;

    const int row = (mbase + lcol < M) ? (mbase + lcol) : (M - 1);
    const unsigned short* arow = A + (size_t)row * EDIM;
    const int koff = lhalf ? 8 : 0;

    v8f acc[4];
    #pragma unroll
    for (int t = 0; t < 4; ++t) {
        const float bv = bias[nblk + t * 16 + lcol];
        #pragma unroll
        for (int r = 0; r < 8; ++r) acc[t][r] = bv;
    }

    #pragma unroll
    for (int s = 0; s < 8; ++s) {
        // A fragment (16-bit A 16x32 layout): this lane needs
        // K = koff+{0..7} and koff+16+{0..7} -> two contiguous 16B loads.
        const unsigned short* ap = arow + s * 32 + koff;
        union { struct { uint4 lo; uint4 hi; } q; v16bf b; } ua;
        ua.q.lo = *(const uint4*)(ap);
        ua.q.hi = *(const uint4*)(ap + 16);

        // All 4 B fragments up front: one ds clause, one wait, 4 WMMAs.
        union { v16u16 u; v16bf b; } ub0, ub1, ub2, ub3;
        ub0.u = *(const v16u16*)&ldsB[0][s][lane][0];
        ub1.u = *(const v16u16*)&ldsB[1][s][lane][0];
        ub2.u = *(const v16u16*)&ldsB[2][s][lane][0];
        ub3.u = *(const v16u16*)&ldsB[3][s][lane][0];

        acc[0] = __builtin_amdgcn_wmma_f32_16x16x32_bf16(
            false, ua.b, false, ub0.b, (short)0, acc[0], false, false);
        acc[1] = __builtin_amdgcn_wmma_f32_16x16x32_bf16(
            false, ua.b, false, ub1.b, (short)0, acc[1], false, false);
        acc[2] = __builtin_amdgcn_wmma_f32_16x16x32_bf16(
            false, ua.b, false, ub2.b, (short)0, acc[2], false, false);
        acc[3] = __builtin_amdgcn_wmma_f32_16x16x32_bf16(
            false, ua.b, false, ub3.b, (short)0, acc[3], false, false);
    }

    // C/D layout: VGPR r, lanes 0-15 -> M=r, lanes 16-31 -> M=r+8.
    #pragma unroll
    for (int t = 0; t < 4; ++t) {
        const int col = nblk + t * 16 + lcol;
        #pragma unroll
        for (int r = 0; r < 8; ++r) {
            const int orow = mbase + r + lhalf * 8;
            if (orow < M) {
                if (OUT_BF16)
                    ((unsigned short*)Cv)[(size_t)orow * N + col] = f2bf(acc[t][r]);
                else
                    ((float*)Cv)[(size_t)orow * N + col] = acc[t][r];
            }
        }
    }
}

// ---------------------------------------------------------------------------
// Softmax + bilinear sampling + weighted sum.
// One wave per (b,q,h); lane = channel (HD=32). Values are bf16 (64 B
// coalesced per corner gather); output written as bf16 to feed final WMMA.
// ---------------------------------------------------------------------------
__device__ __constant__ int c_wh[4] = {100, 50, 25, 13};
__device__ __constant__ int c_st[4] = {0, 10000, 12500, 13125};

__global__ __launch_bounds__(256)
void deform_sample(const float* __restrict__ proj_off,
                   const float* __restrict__ proj_attn,
                   const float* __restrict__ qref,
                   const unsigned short* __restrict__ values,
                   unsigned short* __restrict__ out_acc) {
    const int lane = threadIdx.x & 31;
    const int item = blockIdx.x * 8 + (threadIdx.x >> 5);   // (b*Q+q)*H + h
    const int h  = item & 7;
    const int bq = item >> 3;                               // b*Q + q
    const int b  = bq / QTOT;

    float ix = 0.f, iy = 0.f;
    float logit = -1e30f;
    if (lane < 16) {
        const int   l  = lane >> 2;
        const int   wh = c_wh[l];
        const float rx = qref[(size_t)bq * 2 + 0];
        const float ry = qref[(size_t)bq * 2 + 1];
        const size_t ob = (size_t)bq * 256 + (size_t)((h * 4 + l) * 4 + (lane & 3)) * 2;
        const float ox = proj_off[ob + 0];
        const float oy = proj_off[ob + 1];
        // grid_sample normalization collapses to: pixel = ref*wh + off - 0.5
        ix = rx * (float)wh + ox - 0.5f;
        iy = ry * (float)wh + oy - 0.5f;
        logit = proj_attn[(size_t)bq * 128 + h * 16 + lane];
    }

    // softmax over 16 lanes (xor masks 1/2/4/8 stay within each 16-lane group)
    float m = logit;
    for (int d = 8; d >= 1; d >>= 1) m = fmaxf(m, __shfl_xor(m, d, 32));
    const float e = (lane < 16) ? __expf(logit - m) : 0.f;
    float sum = e;
    for (int d = 8; d >= 1; d >>= 1) sum += __shfl_xor(sum, d, 32);
    const float prob = e / sum;

    float acc = 0.f;
    const unsigned short* vbase = values + (size_t)b * QTOT * 256 + h * 32 + lane;
    #pragma unroll
    for (int j = 0; j < 16; ++j) {
        const float bx = __shfl(ix, j, 32);
        const float by = __shfl(iy, j, 32);
        const float pr = __shfl(prob, j, 32);
        const int l  = j >> 2;
        const int wh = c_wh[l];
        const unsigned short* vl = vbase + (size_t)c_st[l] * 256;

        const float x0f = floorf(bx), y0f = floorf(by);
        const float wx1 = bx - x0f,   wy1 = by - y0f;
        const float wx0 = 1.f - wx1,  wy0 = 1.f - wy1;
        const int x0 = (int)x0f, y0 = (int)y0f;

        float s = 0.f;
        if (y0 >= 0 && y0 < wh) {
            if (x0 >= 0 && x0 < wh)
                s += wy0 * wx0 * bf2f(vl[(size_t)(y0 * wh + x0) * 256]);
            if (x0 + 1 >= 0 && x0 + 1 < wh)
                s += wy0 * wx1 * bf2f(vl[(size_t)(y0 * wh + x0 + 1) * 256]);
        }
        if (y0 + 1 >= 0 && y0 + 1 < wh) {
            if (x0 >= 0 && x0 < wh)
                s += wy1 * wx0 * bf2f(vl[(size_t)((y0 + 1) * wh + x0) * 256]);
            if (x0 + 1 >= 0 && x0 + 1 < wh)
                s += wy1 * wx1 * bf2f(vl[(size_t)((y0 + 1) * wh + x0 + 1) * 256]);
        }
        acc += pr * s;
    }
    // (B,Q,E) with e = h*32 + hd, stored bf16 -> feeds final WMMA A-side
    out_acc[(size_t)bq * 256 + h * 32 + lane] = f2bf(acc);
}

// ---------------------------------------------------------------------------
extern "C" void kernel_launch(void* const* d_in, const int* in_sizes, int n_in,
                              void* d_out, int out_size, void* d_ws, size_t ws_size,
                              hipStream_t stream) {
    (void)in_sizes; (void)n_in; (void)out_size; (void)ws_size;

    const float* queries  = (const float*)d_in[0];
    const float* qref     = (const float*)d_in[1];
    const float* features = (const float*)d_in[2];
    /* d_in[3] = dimensions (int32) — static {100,50,25,13}, baked in */
    const float* W_off    = (const float*)d_in[4];
    const float* b_off    = (const float*)d_in[5];
    const float* W_attn   = (const float*)d_in[6];
    const float* b_attn   = (const float*)d_in[7];
    const float* W_val    = (const float*)d_in[8];
    const float* b_val    = (const float*)d_in[9];
    const float* W_out    = (const float*)d_in[10];
    const float* b_out    = (const float*)d_in[11];
    float* out = (float*)d_out;

    // Workspace layout:
    char* ws = (char*)d_ws;
    const size_t NE = (size_t)MROWS * 256;
    unsigned short* ws_qbf  = (unsigned short*)ws;                 // 13.6 MB
    unsigned short* ws_fbf  = ws_qbf + NE;                         // 13.6 MB
    unsigned short* ws_val  = ws_fbf + NE;                         // 13.6 MB (bf16 values)
    unsigned short* ws_acc  = ws_val + NE;                         // 13.6 MB (bf16 sampled)
    float*          ws_off  = (float*)(ws_acc + NE);               // 27.2 MB
    float*          ws_attn = ws_off + NE;                         // 13.6 MB

    const int n8  = (int)(NE / 8);                  // 850,816
    const int cgx = (n8 + 255) / 256;
    const int mtiles = (MROWS + 15) / 16;           // 1662
    const int gx     = (mtiles + 7) / 8;            // 208
    dim3 blk(256);

    // 0) activation conversions (memory bound, once)
    f32_to_bf16_kernel<<<dim3(cgx), blk, 0, stream>>>(queries,  ws_qbf, n8);
    f32_to_bf16_kernel<<<dim3(cgx), blk, 0, stream>>>(features, ws_fbf, n8);

    // 1) values = features @ W_val + b_val   (bf16 out, feeds gather)
    gemm_bf16_wmma<true ><<<dim3(gx, 256 / 64), blk, 0, stream>>>(ws_fbf, W_val, b_val, ws_val, MROWS, 256);
    // 2) offset projection (f32 out: coordinates need precision)
    gemm_bf16_wmma<false><<<dim3(gx, 256 / 64), blk, 0, stream>>>(ws_qbf, W_off, b_off, ws_off, MROWS, 256);
    // 3) attention-logit projection (f32 out)
    gemm_bf16_wmma<false><<<dim3(gx, 128 / 64), blk, 0, stream>>>(ws_qbf, W_attn, b_attn, ws_attn, MROWS, 128);
    // 4) softmax + bilinear gather + weighted sum (26588 blocks x 8 waves)
    deform_sample<<<dim3(MROWS), blk, 0, stream>>>(ws_off, ws_attn, qref, ws_val, ws_acc);
    // 5) final output projection (f32 out -> d_out)
    gemm_bf16_wmma<false><<<dim3(gx, 256 / 64), blk, 0, stream>>>(ws_acc, W_out, b_out, out, MROWS, 256);
}
